// DecoderLayer_78735340471046
// MI455X (gfx1250) — matrix operand order repaired and checked by reading, not compiled, not run on hardware
//
#include <hip/hip_runtime.h>

// Problem constants (reference: B,T,S,D,H,E,FF = 4,1024,1024,512,8,8,2048)
#define Bc 4
#define Tc 1024
#define Sc 1024
#define Dc 512
#define Hc 8
#define Ec 8
#define FFc 2048
#define HDc 64
#define BTc (Bc * Tc)
#define NEGV (-1e9f)

typedef __attribute__((ext_vector_type(16))) __bf16 v16bf;
typedef __attribute__((ext_vector_type(8))) float v8f;
typedef unsigned int u32x4 __attribute__((ext_vector_type(4)));
typedef int i32x4 __attribute__((ext_vector_type(4)));
typedef int i32x8 __attribute__((ext_vector_type(8)));

__device__ __forceinline__ v8f wmma_bf16(v16bf a, v16bf b, v8f c) {
  // D = A(16x32 bf16) * B(32x16 bf16) + C(16x16 f32)
  return __builtin_amdgcn_wmma_f32_16x16x32_bf16(false, a, false, b, (short)0, c,
                                                 false, false);
}

// A-fragment K index for 16-bit A 16x32 (contiguous order k = 16h + s): vgpr i, lane-half h
__device__ __forceinline__ int a_kofs(int i, int half) {
  return half * 16 + 2 * i;
}

// ---- Tensor Data Mover: 2D tile (tile_d0 elems contiguous x tile_d1 rows) of
// fp32 from global (row stride = stride0 elems) into LDS with row padding of
// 1 DWORD per 32 DWORDs (LDS row pitch = tile_d0 + 1 floats).
__device__ __forceinline__ void tdm_load_2d(unsigned lds_addr, const float* gptr,
                                            unsigned tensor_d0, unsigned tensor_d1,
                                            unsigned tile_d0, unsigned tile_d1,
                                            unsigned long long stride0) {
  unsigned long long ga = (unsigned long long)(uintptr_t)gptr;
  u32x4 g0;
  g0.x = 1u;                                   // count=1, user descriptor
  g0.y = lds_addr;                             // LDS byte address
  g0.z = (unsigned)(ga & 0xffffffffu);         // global_addr[31:0]
  g0.w = (unsigned)((ga >> 32) & 0x1ffffffu) | (2u << 30); // [56:32] | type=2
  i32x8 g1;
  // data_size=2 (4B), pad_enable=1, pad_interval=4 (32 DWORDs), pad_amount=0 (1 DWORD)
  g1[0] = (int)((2u << 16) | (1u << 20) | (4u << 22));
  g1[1] = (int)((tensor_d0 & 0xffffu) << 16);                       // dim0[15:0]
  g1[2] = (int)((tensor_d0 >> 16) | ((tensor_d1 & 0xffffu) << 16)); // dim0[31:16]|dim1[15:0]
  g1[3] = (int)((tensor_d1 >> 16) | (tile_d0 << 16));               // dim1[31:16]|tile0
  g1[4] = (int)(tile_d1 & 0xffffu);                                 // tile1 (tile2=0)
  g1[5] = (int)(stride0 & 0xffffffffull);                           // dim0 stride lo
  g1[6] = (int)((stride0 >> 32) & 0xffffull);                       // stride hi (dim1 stride=0)
  g1[7] = 0;
  i32x4 z4 = {0, 0, 0, 0};
#if __clang_major__ >= 23
  i32x8 z8 = {0, 0, 0, 0, 0, 0, 0, 0};
  __builtin_amdgcn_tensor_load_to_lds(g0, g1, z4, z4, z8, 0);
#else
  __builtin_amdgcn_tensor_load_to_lds(g0, g1, z4, z4, 0);
#endif
}

// ---------------------------------------------------------------- LayerNorm
__global__ void ln_kernel(const float* __restrict__ x, const float* __restrict__ g,
                          const float* __restrict__ b, float* __restrict__ out) {
  int row = blockIdx.x;
  const float* xr = x + (size_t)row * Dc;
  float* orow = out + (size_t)row * Dc;
  int tid = threadIdx.x, lane = tid & 31, w = tid >> 5;
  float s = 0.f, s2 = 0.f;
  for (int i = tid; i < Dc; i += blockDim.x) { float v = xr[i]; s += v; s2 += v * v; }
#pragma unroll
  for (int o = 16; o > 0; o >>= 1) { s += __shfl_xor(s, o, 32); s2 += __shfl_xor(s2, o, 32); }
  __shared__ float sa[8], sb[8];
  if (lane == 0) { sa[w] = s; sb[w] = s2; }
  __syncthreads();
  if (w == 0) {
    s = (lane < 8) ? sa[lane] : 0.f;
    s2 = (lane < 8) ? sb[lane] : 0.f;
#pragma unroll
    for (int o = 4; o > 0; o >>= 1) { s += __shfl_xor(s, o, 32); s2 += __shfl_xor(s2, o, 32); }
    if (lane == 0) { sa[0] = s; sb[0] = s2; }
  }
  __syncthreads();
  float mean = sa[0] / Dc;
  float var = sb[0] / Dc - mean * mean;
  float inv = rsqrtf(var + 1e-5f);
  for (int i = tid; i < Dc; i += blockDim.x)
    orow[i] = (xr[i] - mean) * inv * g[i] + b[i];
}

// --------------------- GEMM (double-buffered TDM): C = A @ W^T + bias (+res)
// A[M,K] fp32 row-major, W[N,K] fp32 row-major. Block = 256 threads (8 waves)
// computing a 128x64 C tile; wave w covers rows 16w..16w+15. Ping-pong LDS
// staging: wave 0 issues TDM loads for stage s+1, waits TENSORcnt<=2 (stage s
// complete; TDM completes in-order), barrier, all waves WMMA stage s from LDS
// while the TDM engine streams stage s+1.
#define LPITCH 33
__global__ void gemm_xwT(const float* __restrict__ A, const float* __restrict__ W,
                         const float* __restrict__ bias,
                         const float* __restrict__ residual, float* __restrict__ C,
                         int M, int N, int K) {
  __shared__ float sA[2][128 * LPITCH];
  __shared__ float sW[2][64 * LPITCH];
  int tid = threadIdx.x;
  int lane = tid & 31;
  int wv = tid >> 5; // 0..7
  int mtile = blockIdx.y * 128;
  int nt = blockIdx.x * 64;
  int mr = lane & 15, half = lane >> 4;
  const float* Abase = A + (size_t)mtile * K;
  const float* Wbase = W + (size_t)nt * K;
  int ns = K / 32;
  if (wv == 0) {
    tdm_load_2d((unsigned)(uintptr_t)(void*)&sA[0][0], Abase, (unsigned)K,
                (unsigned)M, 32u, 128u, (unsigned long long)K);
    tdm_load_2d((unsigned)(uintptr_t)(void*)&sW[0][0], Wbase, (unsigned)K,
                (unsigned)N, 32u, 64u, (unsigned long long)K);
  }
  v8f acc[4] = {};
  for (int s = 0; s < ns; ++s) {
    int cur = s & 1;
    if (wv == 0) {
      if (s + 1 < ns) {
        int nxt = cur ^ 1;
        tdm_load_2d((unsigned)(uintptr_t)(void*)&sA[nxt][0], Abase + (s + 1) * 32,
                    (unsigned)K, (unsigned)M, 32u, 128u, (unsigned long long)K);
        tdm_load_2d((unsigned)(uintptr_t)(void*)&sW[nxt][0], Wbase + (s + 1) * 32,
                    (unsigned)K, (unsigned)N, 32u, 64u, (unsigned long long)K);
        __builtin_amdgcn_s_wait_tensorcnt(2); // stage s landed; s+1 in flight
      } else {
        __builtin_amdgcn_s_wait_tensorcnt(0);
      }
    }
    __syncthreads(); // stage s visible to all waves
    v16bf a;
#pragma unroll
    for (int i = 0; i < 8; ++i) {
      const float* r = &sA[cur][(wv * 16 + mr) * LPITCH + a_kofs(i, half)];
      a[2 * i] = (__bf16)r[0];
      a[2 * i + 1] = (__bf16)r[1];
    }
#pragma unroll
    for (int t = 0; t < 4; ++t) {
      v16bf bf;
#pragma unroll
      for (int j = 0; j < 8; ++j) {
        const float* r = &sW[cur][(t * 16 + mr) * LPITCH + half * 16 + 2 * j];
        bf[2 * j] = (__bf16)r[0];
        bf[2 * j + 1] = (__bf16)r[1];
      }
      acc[t] = wmma_bf16(a, bf, acc[t]);
    }
    __syncthreads(); // all reads of buffer `cur` done before stage s+2 reuses it
  }
  int mt = mtile + wv * 16;
#pragma unroll
  for (int t = 0; t < 4; ++t) {
    int n = nt + 16 * t + mr;
    float bv = bias ? bias[n] : 0.f;
#pragma unroll
    for (int j = 0; j < 8; ++j) {
      int m = mt + j + 8 * half;
      float v = acc[t][j] + bv;
      if (residual) v += residual[(size_t)m * N + n];
      C[(size_t)m * N + n] = v;
    }
  }
}

// ---------------------------------------------------------------- Attention
// One wave per (b, h, 16-row q tile). Scores stripe kept in LDS (16 x 1024 f32).
// q,k,v stored [B, T(S), D] with head h at column offset h*64.
__global__ void attn_kernel(const float* __restrict__ q, const float* __restrict__ k,
                            const float* __restrict__ v, float* __restrict__ out,
                            const unsigned char* __restrict__ amask, // [Tq,Tk] or null
                            const unsigned char* __restrict__ kpad,  // [B,Tk]
                            int Tq, int Tk) {
  __shared__ float sc[16 * 1024];
  int lane = threadIdx.x;
  int qt = blockIdx.x * 16;
  int h = blockIdx.y;
  int b = blockIdx.z;
  int mr = lane & 15, half = lane >> 4;
  const float* qb = q + (size_t)b * Tq * Dc + h * HDc;
  const float* kb = k + (size_t)b * Tk * Dc + h * HDc;
  const float* vb = v + (size_t)b * Tk * Dc + h * HDc;

  v16bf qf[2];
  {
    const float* row = qb + (size_t)(qt + mr) * Dc;
#pragma unroll
    for (int c = 0; c < 2; ++c)
#pragma unroll
      for (int i = 0; i < 8; ++i) {
        int kk = c * 32 + a_kofs(i, half);
        qf[c][2 * i] = (__bf16)row[kk];
        qf[c][2 * i + 1] = (__bf16)row[kk + 1];
      }
  }
  const float scale = 0.125f; // 1/sqrt(64)

  for (int kt = 0; kt < Tk; kt += 16) {
    v8f acc = {};
#pragma unroll
    for (int c = 0; c < 2; ++c) {
      const float* krow = kb + (size_t)(kt + mr) * Dc + c * 32 + half * 16;
      v16bf bf;
#pragma unroll
      for (int j = 0; j < 8; ++j) { bf[2 * j] = (__bf16)krow[2 * j]; bf[2 * j + 1] = (__bf16)krow[2 * j + 1]; }
      acc = wmma_bf16(qf[c], bf, acc);
    }
    int key = kt + mr;
    bool pad = kpad[(size_t)b * Tk + key] != 0;
#pragma unroll
    for (int j = 0; j < 8; ++j) {
      int m = j + 8 * half;
      int qi = qt + m;
      bool dis = pad || (amask && amask[(size_t)qi * Tk + key]);
      sc[m * Tk + key] = dis ? NEGV : acc[j] * scale;
    }
  }
  __syncthreads();

  for (int r = 0; r < 16; ++r) {
    float mx = NEGV;
    for (int i = lane; i < Tk; i += 32) mx = fmaxf(mx, sc[r * Tk + i]);
#pragma unroll
    for (int o = 16; o > 0; o >>= 1) mx = fmaxf(mx, __shfl_xor(mx, o, 32));
    float sum = 0.f;
    for (int i = lane; i < Tk; i += 32) {
      float e = __expf(sc[r * Tk + i] - mx);
      sc[r * Tk + i] = e;
      sum += e;
    }
#pragma unroll
    for (int o = 16; o > 0; o >>= 1) sum += __shfl_xor(sum, o, 32);
    float inv = 1.f / sum;
    for (int i = lane; i < Tk; i += 32) sc[r * Tk + i] *= inv;
  }
  __syncthreads();

  v8f oacc[4] = {};
  for (int kb0 = 0; kb0 < Tk; kb0 += 32) {
    v16bf af;
#pragma unroll
    for (int i = 0; i < 8; ++i) {
      int kk = kb0 + a_kofs(i, half);
      af[2 * i] = (__bf16)sc[mr * Tk + kk];
      af[2 * i + 1] = (__bf16)sc[mr * Tk + kk + 1];
    }
#pragma unroll
    for (int t = 0; t < 4; ++t) {
      v16bf bf;
#pragma unroll
      for (int j = 0; j < 8; ++j) {
        int kk = kb0 + half * 16 + 2 * j;
        bf[2 * j] = (__bf16)vb[(size_t)kk * Dc + t * 16 + mr];
        bf[2 * j + 1] = (__bf16)vb[(size_t)(kk + 1) * Dc + t * 16 + mr];
      }
      oacc[t] = wmma_bf16(af, bf, oacc[t]);
    }
  }
  float* ob = out + (size_t)b * Tq * Dc + h * HDc;
#pragma unroll
  for (int t = 0; t < 4; ++t)
#pragma unroll
    for (int j = 0; j < 8; ++j) {
      int m = qt + j + 8 * half;
      ob[(size_t)m * Dc + t * 16 + mr] = oacc[t][j];
    }
}

// ---------------------------------------------------------------- Router
__global__ void router_kernel(const float* __restrict__ x, const float* __restrict__ rw,
                              const float* __restrict__ rb, float* __restrict__ gate,
                              int* __restrict__ eidx, int* __restrict__ counts) {
  int tok = (blockIdx.x * blockDim.x + threadIdx.x) >> 5;
  int lane = threadIdx.x & 31;
  if (tok >= BTc) return;
  const float* xr = x + (size_t)tok * Dc;
  float logits[Ec];
#pragma unroll
  for (int e = 0; e < Ec; ++e) {
    const float* w = rw + (size_t)e * Dc;
    float s = 0.f;
    for (int d = lane; d < Dc; d += 32) s += xr[d] * w[d];
#pragma unroll
    for (int o = 16; o > 0; o >>= 1) s += __shfl_xor(s, o, 32);
    logits[e] = s + rb[e];
  }
  float mx = logits[0];
  int am = 0;
#pragma unroll
  for (int e = 1; e < Ec; ++e)
    if (logits[e] > mx) { mx = logits[e]; am = e; }
  float sum = 0.f;
#pragma unroll
  for (int e = 0; e < Ec; ++e) sum += __expf(logits[e] - mx);
  if (lane == 0) {
    gate[tok] = 1.f / sum; // exp(mx-mx)/sum
    eidx[tok] = am;
    atomicAdd(&counts[am], 1);
  }
}

__global__ void scan_kernel(const int* __restrict__ counts, int* __restrict__ expoff) {
  if (threadIdx.x == 0 && blockIdx.x == 0) {
    int acc = 0;
    for (int e = 0; e < Ec; ++e) { expoff[e] = acc; acc += counts[e]; }
    expoff[Ec] = acc;
  }
}

__global__ void scatter_kernel(const int* __restrict__ eidx, const int* __restrict__ expoff,
                               int* __restrict__ cursor, int* __restrict__ perm) {
  int t = blockIdx.x * blockDim.x + threadIdx.x;
  if (t >= BTc) return;
  int e = eidx[t];
  int p = atomicAdd(&cursor[e], 1);
  perm[expoff[e] + p] = t;
}

// ----------------------------------------- MoE GEMM1: h = relu(x_g @ w1[e] + b1[e])
// grid = (FF/64, BT/16, E), block = 32
__global__ void moe_gemm1(const float* __restrict__ X, const float* __restrict__ w1,
                          const float* __restrict__ b1, const int* __restrict__ perm,
                          const int* __restrict__ expoff, float* __restrict__ Hbuf) {
  int lane = threadIdx.x;
  int e = blockIdx.z;
  int seg = expoff[e], cnt = expoff[e + 1] - seg;
  int mt = blockIdx.y * 16;
  if (mt >= cnt) return;
  int nt = blockIdx.x * 64;
  int mr = lane & 15, half = lane >> 4;
  bool rv = (mt + mr) < cnt;
  const float* xrow = X + (size_t)(rv ? perm[seg + mt + mr] : 0) * Dc;
  const float* W = w1 + (size_t)e * Dc * FFc; // [D,FF]
  v8f acc[4] = {};
  for (int k0 = 0; k0 < Dc; k0 += 32) {
    v16bf a;
#pragma unroll
    for (int i = 0; i < 8; ++i) {
      int kk = k0 + a_kofs(i, half);
      a[2 * i] = (__bf16)(rv ? xrow[kk] : 0.f);
      a[2 * i + 1] = (__bf16)(rv ? xrow[kk + 1] : 0.f);
    }
#pragma unroll
    for (int t = 0; t < 4; ++t) {
      int n = nt + 16 * t + mr;
      v16bf bf;
#pragma unroll
      for (int j = 0; j < 8; ++j) {
        int kk = k0 + half * 16 + 2 * j;
        bf[2 * j] = (__bf16)W[(size_t)kk * FFc + n];
        bf[2 * j + 1] = (__bf16)W[(size_t)(kk + 1) * FFc + n];
      }
      acc[t] = wmma_bf16(a, bf, acc[t]);
    }
  }
  const float* be = b1 + (size_t)e * FFc;
#pragma unroll
  for (int t = 0; t < 4; ++t) {
    int n = nt + 16 * t + mr;
    float bv = be[n];
#pragma unroll
    for (int j = 0; j < 8; ++j) {
      int m = mt + j + 8 * half;
      if (m < cnt) Hbuf[(size_t)(seg + m) * FFc + n] = fmaxf(acc[t][j] + bv, 0.f);
    }
  }
}

// ------------- MoE GEMM2: out[tok] = x2[tok] + gate*mask*(h_g @ w2[e] + b2[e])
// grid = (D/64, BT/16, E), block = 32
__global__ void moe_gemm2(const float* __restrict__ Hbuf, const float* __restrict__ w2,
                          const float* __restrict__ b2, const int* __restrict__ perm,
                          const int* __restrict__ expoff, const float* __restrict__ gate,
                          const unsigned char* __restrict__ tmask,
                          const float* __restrict__ x2, float* __restrict__ Out) {
  int lane = threadIdx.x;
  int e = blockIdx.z;
  int seg = expoff[e], cnt = expoff[e + 1] - seg;
  int mt = blockIdx.y * 16;
  if (mt >= cnt) return;
  int nt = blockIdx.x * 64;
  int mr = lane & 15, half = lane >> 4;
  bool rv = (mt + mr) < cnt;
  const float* arow = Hbuf + (size_t)(rv ? (seg + mt + mr) : seg) * FFc;
  const float* W = w2 + (size_t)e * FFc * Dc; // [FF,D]
  v8f acc[4] = {};
  for (int k0 = 0; k0 < FFc; k0 += 32) {
    v16bf a;
#pragma unroll
    for (int i = 0; i < 8; ++i) {
      int kk = k0 + a_kofs(i, half);
      a[2 * i] = (__bf16)(rv ? arow[kk] : 0.f);
      a[2 * i + 1] = (__bf16)(rv ? arow[kk + 1] : 0.f);
    }
#pragma unroll
    for (int t = 0; t < 4; ++t) {
      int n = nt + 16 * t + mr;
      v16bf bf;
#pragma unroll
      for (int j = 0; j < 8; ++j) {
        int kk = k0 + half * 16 + 2 * j;
        bf[2 * j] = (__bf16)W[(size_t)kk * Dc + n];
        bf[2 * j + 1] = (__bf16)W[(size_t)(kk + 1) * Dc + n];
      }
      acc[t] = wmma_bf16(a, bf, acc[t]);
    }
  }
  const float* be = b2 + (size_t)e * Dc;
#pragma unroll
  for (int t = 0; t < 4; ++t) {
    int n = nt + 16 * t + mr;
    float bv = be[n];
#pragma unroll
    for (int j = 0; j < 8; ++j) {
      int m = mt + j + 8 * half;
      if (m < cnt) {
        int tok = perm[seg + m];
        float sc = gate[tok] * (tmask[tok] ? 1.f : 0.f);
        Out[(size_t)tok * Dc + n] = x2[(size_t)tok * Dc + n] + (acc[t][j] + bv) * sc;
      }
    }
  }
}

// ---------------------------------------------------------------- launch
extern "C" void kernel_launch(void* const* d_in, const int* in_sizes, int n_in,
                              void* d_out, int out_size, void* d_ws, size_t ws_size,
                              hipStream_t stream) {
  const float* tgt = (const float*)d_in[0];
  const float* src = (const float*)d_in[1];
  const unsigned char* tgt_mask = (const unsigned char*)d_in[2];
  const unsigned char* tgt_pad = (const unsigned char*)d_in[3];
  const unsigned char* src_pad = (const unsigned char*)d_in[4];
  const unsigned char* tok_mask = (const unsigned char*)d_in[5];
  const float* ln1g = (const float*)d_in[7];
  const float* ln1b = (const float*)d_in[8];
  const float* ln2g = (const float*)d_in[9];
  const float* ln2b = (const float*)d_in[10];
  const float* ln3g = (const float*)d_in[11];
  const float* ln3b = (const float*)d_in[12];
  const float* sa_win = (const float*)d_in[13];
  const float* sa_bin = (const float*)d_in[14];
  const float* sa_wo = (const float*)d_in[15];
  const float* sa_bo = (const float*)d_in[16];
  const float* ca_win = (const float*)d_in[17];
  const float* ca_bin = (const float*)d_in[18];
  const float* ca_wo = (const float*)d_in[19];
  const float* ca_bo = (const float*)d_in[20];
  const float* router_w = (const float*)d_in[21];
  const float* router_b = (const float*)d_in[22];
  const float* w1 = (const float*)d_in[23];
  const float* b1 = (const float*)d_in[24];
  const float* w2 = (const float*)d_in[25];
  const float* b2 = (const float*)d_in[26];
  float* out = (float*)d_out;

  const size_t NTD = (size_t)BTc * Dc;
  float* fw = (float*)d_ws;
  float* tb = fw;             // LN outputs (reused)
  float* qb = tb + NTD;
  float* kb = qb + NTD;
  float* vb = kb + NTD;
  float* ab = vb + NTD;       // attention output
  float* x1 = ab + NTD;
  float* x2 = x1 + NTD;
  float* hb = x2 + NTD;       // MoE hidden [BT, FF]
  float* gate = hb + (size_t)BTc * FFc;
  int* eidx = (int*)(gate + BTc);
  int* counts = eidx + BTc;
  int* cursor = counts + Ec;
  int* expoff = cursor + Ec;
  int* perm = expoff + (Ec + 1);

  dim3 gD(Dc / 64, BTc / 128);     // 128x64 tiles, 8-wave blocks
  dim3 gA(Tc / 16, Hc, Bc);        // attention blocks
  const size_t DD = (size_t)Dc * Dc;

  // ---- self attention ----
  ln_kernel<<<BTc, 256, 0, stream>>>(tgt, ln1g, ln1b, tb);
  gemm_xwT<<<gD, 256, 0, stream>>>(tb, sa_win, sa_bin, nullptr, qb, BTc, Dc, Dc);
  gemm_xwT<<<gD, 256, 0, stream>>>(tb, sa_win + DD, sa_bin + Dc, nullptr, kb, BTc, Dc, Dc);
  gemm_xwT<<<gD, 256, 0, stream>>>(tb, sa_win + 2 * DD, sa_bin + 2 * Dc, nullptr, vb, BTc, Dc, Dc);
  attn_kernel<<<gA, 32, 0, stream>>>(qb, kb, vb, ab, tgt_mask, tgt_pad, Tc, Tc);
  gemm_xwT<<<gD, 256, 0, stream>>>(ab, sa_wo, sa_bo, tgt, x1, BTc, Dc, Dc);

  // ---- cross attention ----
  ln_kernel<<<BTc, 256, 0, stream>>>(x1, ln2g, ln2b, tb);
  gemm_xwT<<<gD, 256, 0, stream>>>(tb, ca_win, ca_bin, nullptr, qb, BTc, Dc, Dc);
  gemm_xwT<<<gD, 256, 0, stream>>>(src, ca_win + DD, ca_bin + Dc, nullptr, kb, Bc * Sc, Dc, Dc);
  gemm_xwT<<<gD, 256, 0, stream>>>(src, ca_win + 2 * DD, ca_bin + 2 * Dc, nullptr, vb, Bc * Sc, Dc, Dc);
  attn_kernel<<<gA, 32, 0, stream>>>(qb, kb, vb, ab, nullptr, src_pad, Tc, Sc);
  gemm_xwT<<<gD, 256, 0, stream>>>(ab, ca_wo, ca_bo, x1, x2, BTc, Dc, Dc);

  // ---- switch MoE (top-1, gather per expert) ----
  ln_kernel<<<BTc, 256, 0, stream>>>(x2, ln3g, ln3b, tb);
  hipMemsetAsync(counts, 0, 2 * Ec * sizeof(int), stream); // counts + cursor
  router_kernel<<<BTc / 8, 256, 0, stream>>>(tb, router_w, router_b, gate, eidx, counts);
  scan_kernel<<<1, 1, 0, stream>>>(counts, expoff);
  scatter_kernel<<<BTc / 256, 256, 0, stream>>>(eidx, expoff, cursor, perm);
  moe_gemm1<<<dim3(FFc / 64, BTc / 16, Ec), 32, 0, stream>>>(tb, w1, b1, perm, expoff, hb);
  moe_gemm2<<<dim3(Dc / 64, BTc / 16, Ec), 32, 0, stream>>>(hb, w2, b2, perm, expoff, gate,
                                                            tok_mask, x2, out);
  (void)in_sizes; (void)n_in; (void)out_size; (void)ws_size;
}
